// VectorQuantizer_67645734912598
// MI455X (gfx1250) — matrix operand order repaired and checked by reading, not compiled, not run on hardware
//
#include <hip/hip_runtime.h>
#include <hip/hip_bf16.h>
#include <stdint.h>

// ---- WMMA vector types (gfx1250, wave32) ----
typedef __attribute__((ext_vector_type(16))) __bf16        v16bf;
typedef __attribute__((ext_vector_type(8)))  float         v8f;
typedef __attribute__((ext_vector_type(16))) unsigned short v16u;
typedef __attribute__((ext_vector_type(8)))  unsigned short v8u;

#define NUM_EMB     512
#define DIM         64
#define NROWS       65536              // 64*32*32
#define Q_ELEMS     (NROWS * DIM)      // 4194304
#define O_LOSS      0
#define O_Q         1
#define O_P         (1 + Q_ELEMS)
#define O_IDX       (2 + Q_ELEMS)
#define CB_STRIDE   72                 // padded LDS row stride in bf16 elems (144B -> spreads banks)
#define WG_THREADS  256
#define ROWS_PER_WG 128                // 8 waves * 16 rows
#define SMEM_BYTES  (NUM_EMB * 4 + NUM_EMB * CB_STRIDE * 2)  // 2048 + 73728 = 75776

// float -> bf16 bits, round-to-nearest-even (matches HW convert semantics)
__device__ __forceinline__ unsigned short f2bf(float f) {
  unsigned u = __float_as_uint(f);
  u += 0x7FFFu + ((u >> 16) & 1u);
  return (unsigned short)(u >> 16);
}

__global__ void __launch_bounds__(512) vq_zero(float* err, unsigned* hist) {
  if (threadIdx.x == 0) *err = 0.0f;
  if (threadIdx.x < NUM_EMB) hist[threadIdx.x] = 0u;
}

__global__ void __launch_bounds__(WG_THREADS) vq_main(
    const float* __restrict__ x, const float* __restrict__ cb,
    float* __restrict__ out, float* __restrict__ err_acc,
    unsigned* __restrict__ hist) {
  extern __shared__ char smem[];
  float*          s_cnorm = (float*)smem;                           // 512 f32
  unsigned short* s_cb    = (unsigned short*)(smem + NUM_EMB * 4);  // 512 x CB_STRIDE bf16

  const int tid = threadIdx.x;

  // ---- stage full codebook into LDS as bf16; compute ||c_k||^2 in f32 ----
  for (int row = tid; row < NUM_EMB; row += WG_THREADS) {
    const float4*   src = (const float4*)(cb + row * DIM);
    unsigned short* dst = s_cb + row * CB_STRIDE;
    float ss = 0.0f;
#pragma unroll
    for (int v = 0; v < DIM / 4; ++v) {
      float4 f = src[v];
      ss += f.x * f.x + f.y * f.y + f.z * f.z + f.w * f.w;
      dst[4 * v + 0] = f2bf(f.x);
      dst[4 * v + 1] = f2bf(f.y);
      dst[4 * v + 2] = f2bf(f.z);
      dst[4 * v + 3] = f2bf(f.w);
    }
    s_cnorm[row] = ss;
  }
  __syncthreads();

  const int wave    = tid >> 5;
  const int lane    = tid & 31;
  const int half    = lane >> 4;   // 0: codes M=0..7 of tile, 1: M=8..15
  const int l16     = lane & 15;   // activation row within wave tile (B column N)
  const int rowbase = blockIdx.x * ROWS_PER_WG + wave * 16;

  // ---- B fragments: x tile (K x N = dims x rows), both 32-dim chunks.
  // B k order chosen here (contiguous, same as A): lanes 0-15 hold K = 0..15, lanes 16-31 hold K = 16..31 (N = lane%16).
  v16bf b0, b1;
  {
    const float* xr = x + (size_t)(rowbase + l16) * DIM + half * 16;
    v16u u0, u1;
#pragma unroll
    for (int e = 0; e < 16; ++e) {
      u0[e] = f2bf(xr[e]);        // dims [half*16 .. half*16+15]
      u1[e] = f2bf(xr[32 + e]);   // dims [32+half*16 .. 32+half*16+15]
    }
    b0 = __builtin_bit_cast(v16bf, u0);
    b1 = __builtin_bit_cast(v16bf, u1);
  }

  // ---- loop over 32 code tiles: S = C_tile(16xK) * X^T(Kx16), K=64 via 2 WMMAs ----
  float best = 3.4e38f;
  int   bidx = 0;
  for (int ct = 0; ct < 32; ++ct) {
    // A 16-bit 16x32 layout (matching B): lane<16 -> K {0..15}; lane>=16 -> K {16..31}
    const unsigned short* ap = s_cb + (ct * 16 + l16) * CB_STRIDE;
    const int ab = half * 16;
    v8u a0lo = *(const v8u*)(ap + ab);            // chunk0 first 8 K
    v8u a0hi = *(const v8u*)(ap + ab + 8);        // chunk0 second 8 K
    v8u a1lo = *(const v8u*)(ap + 32 + ab);       // chunk1
    v8u a1hi = *(const v8u*)(ap + 32 + ab + 8);
    v16u ua0, ua1;
#pragma unroll
    for (int e = 0; e < 8; ++e) {
      ua0[e] = a0lo[e]; ua0[8 + e] = a0hi[e];
      ua1[e] = a1lo[e]; ua1[8 + e] = a1hi[e];
    }
    v16bf a0 = __builtin_bit_cast(v16bf, ua0);
    v16bf a1 = __builtin_bit_cast(v16bf, ua1);

    v8f c = {};
    c = __builtin_amdgcn_wmma_f32_16x16x32_bf16(false, a0, false, b0, (short)0, c, false, false);
    c = __builtin_amdgcn_wmma_f32_16x16x32_bf16(false, a1, false, b1, (short)0, c, false, false);

    // score = ||c||^2 - 2 x.c  (||x||^2 constant per row -> irrelevant to argmin)
    const int    cbase = ct * 16 + half * 8;
    const float4* cn   = (const float4*)(s_cnorm + cbase);
    float4 n0 = cn[0], n1 = cn[1];
    float sc;
    sc = n0.x - 2.0f * c[0]; if (sc < best) { best = sc; bidx = cbase + 0; }
    sc = n0.y - 2.0f * c[1]; if (sc < best) { best = sc; bidx = cbase + 1; }
    sc = n0.z - 2.0f * c[2]; if (sc < best) { best = sc; bidx = cbase + 2; }
    sc = n0.w - 2.0f * c[3]; if (sc < best) { best = sc; bidx = cbase + 3; }
    sc = n1.x - 2.0f * c[4]; if (sc < best) { best = sc; bidx = cbase + 4; }
    sc = n1.y - 2.0f * c[5]; if (sc < best) { best = sc; bidx = cbase + 5; }
    sc = n1.z - 2.0f * c[6]; if (sc < best) { best = sc; bidx = cbase + 6; }
    sc = n1.w - 2.0f * c[7]; if (sc < best) { best = sc; bidx = cbase + 7; }
  }

  // ---- merge the two half-wave code subsets (lane l <-> l^16 hold same row) ----
  {
    float obest = __shfl_xor(best, 16, 32);
    int   oidx  = __shfl_xor(bidx, 16, 32);
    if (obest < best || (obest == best && oidx < bidx)) { best = obest; bidx = oidx; }
  }

  // ---- indices (+ histogram): one writer per row ----
  if (half == 0) {
    out[(size_t)O_IDX + rowbase + l16] = (float)bidx;
    atomicAdd(&hist[bidx], 1u);
  }

  // ---- quantized_st (exact f32 gather) + squared-error accumulation ----
  float errsum = 0.0f;
#pragma unroll
  for (int r = 0; r < 16; ++r) {
    int id = __shfl(bidx, r, 32);
    const float2 cv = ((const float2*)(cb + (size_t)id * DIM))[lane];
    const float2 xv = ((const float2*)(x + (size_t)(rowbase + r) * DIM))[lane];
    float* q = out + (size_t)O_Q + (size_t)(rowbase + r) * DIM + lane * 2;
    q[0] = cv.x;
    q[1] = cv.y;
    float dx = cv.x - xv.x, dy = cv.y - xv.y;
    errsum += dx * dx + dy * dy;
  }
#pragma unroll
  for (int s = 16; s > 0; s >>= 1) errsum += __shfl_xor(errsum, s, 32);
  if (lane == 0) atomicAdd(err_acc, errsum);
}

__global__ void __launch_bounds__(512) vq_final(const float* __restrict__ err_acc,
                                                const unsigned* __restrict__ hist,
                                                float* __restrict__ out) {
  __shared__ float red[512];
  const int t = threadIdx.x;
  float p = (float)hist[t] * (1.0f / (float)NROWS);
  red[t] = -p * logf(p + 1e-10f);
  __syncthreads();
#pragma unroll
  for (int s = 256; s > 0; s >>= 1) {
    if (t < s) red[t] += red[t + s];
    __syncthreads();
  }
  if (t == 0) {
    out[O_P]    = expf(red[0]);
    out[O_LOSS] = 1.25f * (*err_acc) / (float)Q_ELEMS;  // q_loss + 0.25*e_loss, both equal fwd
  }
}

extern "C" void kernel_launch(void* const* d_in, const int* in_sizes, int n_in,
                              void* d_out, int out_size, void* d_ws, size_t ws_size,
                              hipStream_t stream) {
  const float* x   = (const float*)d_in[0];   // (64,32,32,64) f32
  const float* cb  = (const float*)d_in[1];   // (512,64) f32
  float*    out    = (float*)d_out;
  float*    err    = (float*)d_ws;                       // 1 f32 @ 0
  unsigned* hist   = (unsigned*)((char*)d_ws + 16);      // 512 u32 @ 16

  hipLaunchKernelGGL(vq_zero, dim3(1), dim3(512), 0, stream, err, hist);
  hipLaunchKernelGGL(vq_main, dim3(NROWS / ROWS_PER_WG), dim3(WG_THREADS),
                     SMEM_BYTES, stream, x, cb, out, err, hist);
  hipLaunchKernelGGL(vq_final, dim3(1), dim3(512), 0, stream, err, hist, out);
}
